// ModulatedConv2d_80229989090057
// MI455X (gfx1250) — compile-verified, matrix operand order checked by reading, not run on hardware
//
#include <hip/hip_runtime.h>
#include <hip/hip_bf16.h>
#include <math.h>

// ---------------------------------------------------------------------------
// ModulatedConv2d (StyleGAN2): B=16, CIN=COUT=128, K=3, SDIM=512, H=W=128
// Implicit GEMM on v_wmma_f32_16x16x32_bf16 (wave32, gfx1250).
// Big-workspace path: bf16 NHWC-transposed x staged into LDS by the Tensor
// Data Mover (tensor_load_to_lds + s_wait_tensorcnt), with TDM pad insertion
// reproducing the bank-spread LDS layout. Fallback: VGPR staging.
// ---------------------------------------------------------------------------

#define BATCH 16
#define CIN   128
#define COUT  128
#define KS    3
#define SDIM  512
#define HH    128
#define WW    128

#define KTOT   (CIN * KS * KS)   // 1152 = GEMM K dimension
#define NCHUNK (KTOT / 32)       // 36 K-chunks of 32
#define MT     (COUT / 16)       // 8 M-tiles

// conv-kernel tiling: one block = (b, h, 64-col segment)
#define SEGW   64
#define CPAD   136               // cin padding in halves (256B data + 16B pad)
#define WPADN  66                // 64 cols + 2 halo
#define LDSN   (3 * WPADN * CPAD) // 26928 halves = 53856 B

typedef __attribute__((ext_vector_type(16))) __bf16 v16bf;
typedef __attribute__((ext_vector_type(8)))  __bf16 v8bf;
typedef __attribute__((ext_vector_type(8)))  float  v8f;
typedef __attribute__((ext_vector_type(4)))  unsigned int u32x4;
typedef __attribute__((ext_vector_type(8)))  int i32x8;
typedef __attribute__((ext_vector_type(4)))  int i32x4;

__device__ __forceinline__ float ldg(const float* p) { return __builtin_nontemporal_load(p); }

static __device__ const float kScale = 0.029462782549439476f; // 1/sqrt(1152)

// ---------------------------------------------------------------------------
// Kernel 1: s[b][cin] = style[b] . mod_w[cin]
// ---------------------------------------------------------------------------
__global__ void k_style(const float* __restrict__ style,
                        const float* __restrict__ mod_w,
                        float* __restrict__ s) {
    int b = blockIdx.x;
    int c = threadIdx.x;                       // 128 threads
    const float* sp = style + b * SDIM;
    const float* mp = mod_w + c * SDIM;
    float acc = 0.f;
    for (int k = 0; k < SDIM; ++k) acc += sp[k] * mp[k];
    s[b * CIN + c] = acc;
}

// ---------------------------------------------------------------------------
// Kernel 2: demod[b][cout] = rsqrt( scale^2 * sum_{cin,t}(w*s)^2 + 1e-8 )
// ---------------------------------------------------------------------------
__global__ void k_demod(const float* __restrict__ weight,
                        const float* __restrict__ s,
                        float* __restrict__ demod) {
    int bc   = blockIdx.x;
    int b    = bc / COUT;
    int cout = bc % COUT;
    const float* wp = weight + (size_t)cout * KTOT;
    const float* sp = s + b * CIN;
    float acc = 0.f;
    for (int i = threadIdx.x; i < KTOT; i += 128) {
        int cin = i / 9;
        float v = wp[i] * sp[cin];
        acc += v * v;
    }
    __shared__ float red[128];
    red[threadIdx.x] = acc;
    __syncthreads();
    for (int off = 64; off > 0; off >>= 1) {
        if (threadIdx.x < off) red[threadIdx.x] += red[threadIdx.x + off];
        __syncthreads();
    }
    if (threadIdx.x == 0)
        demod[bc] = rsqrtf(kScale * kScale * red[0] + 1e-8f);
}

// ---------------------------------------------------------------------------
// Kernel 3: pack modulated weights into WMMA A-fragment lane layout (bf16).
// K ordering: k = tap*128 + cin.  Per-lane 16 halves contiguous (32B).
// ---------------------------------------------------------------------------
__global__ void k_pack(const float* __restrict__ weight,
                       const float* __restrict__ s,
                       const float* __restrict__ demod,
                       __bf16* __restrict__ wsW) {
    int blk = blockIdx.x;                // B*MT*NCHUNK = 4608
    int kc  = blk % NCHUNK;
    int mt  = (blk / NCHUNK) % MT;
    int b   = blk / (NCHUNK * MT);
    int l   = threadIdx.x;               // 32 lanes
    int hi  = l >> 4;
    int cout = mt * 16 + (l & 15);

    int tap = kc >> 2;                   // 0..8
    int kh  = tap / 3, kw = tap % 3;
    int cinbase = (kc & 3) * 32;

    float dm = demod[b * COUT + cout];
    const float* sp = s + b * CIN;
    const float* wp = weight + ((size_t)cout * CIN) * 9 + kh * 3 + kw;

    __bf16 frag[16];
#pragma unroll
    for (int e = 0; e < 16; ++e) {
        int Krel = (e & 7) + ((e >> 3) << 4) + hi * 8;
        int cin  = cinbase + Krel;
        float v  = kScale * wp[(size_t)cin * 9] * sp[cin] * dm;
        frag[e]  = (__bf16)v;
    }
    __bf16* dst = wsW + ((((size_t)b * MT + mt) * NCHUNK + kc) * 32 + l) * 16;
    *(v16bf*)dst = *(const v16bf*)frag;
}

// ---------------------------------------------------------------------------
// Kernel 3b (big-ws path): x (B,CIN,H,W) f32 -> xbf (B,H,W,CIN) bf16.
// One block per (b,h); transpose through LDS so both sides are coalesced.
// ---------------------------------------------------------------------------
__global__ void k_cvt(const float* __restrict__ x, __bf16* __restrict__ xbf) {
    __shared__ __bf16 t[CIN * 132];
    int b = blockIdx.x >> 7;
    int h = blockIdx.x & (HH - 1);
    for (int i = threadIdx.x; i < CIN * WW; i += 256) {
        int w   = i & (WW - 1);
        int cin = i >> 7;
        t[cin * 132 + w] = (__bf16)ldg(x + (((size_t)b * CIN + cin) * HH + h) * WW + w);
    }
    __syncthreads();
    __bf16* dst = xbf + (((size_t)b * HH + h) * WW) * CIN;
    for (int i = threadIdx.x; i < CIN * WW; i += 256) {
        int cin = i & (CIN - 1);
        int w   = i >> 7;
        dst[(size_t)w * CIN + cin] = t[cin * 132 + w];
    }
}

// ---------------------------------------------------------------------------
// Kernel 4: implicit-GEMM conv + noise epilogue.
// MODE 0: stage f32 x via VGPRs (safe fallback).
// MODE 1: stage bf16 NHWC x via TDM (tensor_load_to_lds), 65 w-rows of 256B
//         with 16B pad after each -> LDS [r][wpad][cin padded to 136].
// ---------------------------------------------------------------------------
template <int MODE>
__global__ void __launch_bounds__(256, 2)
k_conv(const float* __restrict__ x,
       const __bf16* __restrict__ xbf,
       const float* __restrict__ noise,
       const float* __restrict__ noise_w,
       const __bf16* __restrict__ wsW,
       float* __restrict__ out) {
    __shared__ __align__(16) __bf16 xs[LDSN];

    int blk  = blockIdx.x;
    int wseg = blk & 1;
    int h    = (blk >> 1) & (HH - 1);
    int b    = blk >> 8;
    int wbase = wseg * SEGW;

    int tid  = threadIdx.x;
    int lane = tid & 31;
    int mt   = tid >> 5;                 // wave id = M-tile
    int hi   = lane >> 4;
    int col  = lane & 15;

    if (MODE == 0) {
        // ---- VGPR staging: 3 rows x 66 padded cols x 128 cin ----
        const int total = 3 * CIN * WPADN;   // 25344
        for (int i = tid; i < total; i += 256) {
            int wpad = i % WPADN;
            int cin  = (i / WPADN) & (CIN - 1);
            int r    = i / (WPADN * CIN);
            int wg   = wbase - 1 + wpad;
            int row  = h - 1 + r;
            float v = 0.f;
            if ((unsigned)wg < WW && (unsigned)row < HH)
                v = ldg(x + (((size_t)b * CIN + cin) * HH + row) * WW + wg);
            xs[(r * WPADN + wpad) * CPAD + cin] = (__bf16)v;
        }
    } else {
        // ---- TDM staging ----
        // zero out-of-image rows (only h==0 / h==127)
        if (h == 0 || h == HH - 1) {
            int r = (h == 0) ? 0 : 2;
            for (int i = tid; i < WPADN * (CIN / 4); i += 256) {
                int wpad = i / (CIN / 4);
                int c4   = (i % (CIN / 4)) * 4;
                *(unsigned long long*)(xs + (r * WPADN + wpad) * CPAD + c4) = 0ull;
            }
        }
        // zero the one out-of-image halo column of this segment
        {
            int wpadZ = (wbase == 0) ? 0 : (WPADN - 1);
            for (int i = tid; i < 3 * (CIN / 4); i += 256) {
                int r  = i / (CIN / 4);
                int c4 = (i % (CIN / 4)) * 4;
                *(unsigned long long*)(xs + (r * WPADN + wpadZ) * CPAD + c4) = 0ull;
            }
        }
        if (tid < 32) {
            int w0    = (wbase == 0) ? 0 : (wbase - 1);   // first valid global w
            int wpad0 = (wbase == 0) ? 1 : 0;             // its LDS wpad slot
            unsigned ldsbase = (unsigned)(uintptr_t)xs;
            // 65 w-rows * 256B, data_size=8B -> 2080 elements; pad 4 DW after
            // every 64 DW => LDS row pitch 272B = CPAD halves.
            const int g1w0 = (3 << 16) | (1 << 20) | (5 << 22) | (3 << 25);
#pragma unroll
            for (int r = 0; r < 3; ++r) {
                int row = h - 1 + r;
                if ((unsigned)row < HH) {
                    const __bf16* gp = xbf + ((((size_t)b * HH + row) * WW) + w0) * CIN;
                    unsigned lds_addr = ldsbase + (unsigned)((r * WPADN + wpad0) * CPAD) * 2u;
                    unsigned long long ga = (unsigned long long)(uintptr_t)gp;
                    u32x4 g0 = { 1u,                                   // count=1
                                 lds_addr,
                                 (unsigned)(ga & 0xffffffffu),
                                 (unsigned)((ga >> 32) & 0x01ffffffu) | (2u << 30) }; // type=2
                    i32x8 g1 = { g1w0,
                                 (int)(2080u << 16),    // tensor_dim0[15:0]
                                 (int)(1u << 16),       // tensor_dim1 = 1
                                 (int)(2080u << 16),    // tile_dim0
                                 1,                     // tile_dim1 = 1
                                 2080,                  // tensor_dim0_stride
                                 0, 0 };
                    i32x4 z4 = { 0, 0, 0, 0 };
                    i32x8 z8 = { 0, 0, 0, 0, 0, 0, 0, 0 };
                    __builtin_amdgcn_tensor_load_to_lds(g0, g1, z4, z4, z8, 0);
                }
            }
            __builtin_amdgcn_s_wait_tensorcnt(0);
        }
    }
    __syncthreads();

    // ---- main GEMM loop: 36 K-chunks, 4 N-tiles per wave ----
    v8f acc[4];
#pragma unroll
    for (int nt = 0; nt < 4; ++nt) acc[nt] = (v8f)0.f;

    const __bf16* aBase = wsW + (((size_t)b * MT + mt) * NCHUNK) * 512 + lane * 16;

    for (int kc = 0; kc < NCHUNK; ++kc) {
        int tap = kc >> 2;
        int kh  = tap / 3, kw = tap % 3;
        int cin_lo = (kc & 3) * 32 + hi * 8;

        v16bf a = *(const v16bf*)(aBase + (size_t)kc * 512);

#pragma unroll
        for (int nt = 0; nt < 4; ++nt) {
            int n_local = nt * 16 + col;
            int addr = (kh * WPADN + n_local + kw) * CPAD + cin_lo;
            v8bf g0 = *(const v8bf*)(xs + addr);        // K 0..7   (rel)
            v8bf g1 = *(const v8bf*)(xs + addr + 16);   // K 16..23 (rel)
            v16bf bm;
#pragma unroll
            for (int e = 0; e < 8; ++e) { bm[e] = g0[e]; bm[e + 8] = g1[e]; }
            acc[nt] = __builtin_amdgcn_wmma_f32_16x16x32_bf16(
                false, a, false, bm, (short)0, acc[nt], false, false);
        }
    }

    // ---- epilogue: out = conv + noise_weight[cout] * noise ----
    float nwv[8];
#pragma unroll
    for (int r = 0; r < 8; ++r) nwv[r] = noise_w[mt * 16 + hi * 8 + r];

#pragma unroll
    for (int nt = 0; nt < 4; ++nt) {
        int w  = wbase + nt * 16 + col;
        float nz = noise[((size_t)b * HH + h) * WW + w];
#pragma unroll
        for (int r = 0; r < 8; ++r) {
            int cout = mt * 16 + hi * 8 + r;
            out[(((size_t)b * COUT + cout) * HH + h) * WW + w] = acc[nt][r] + nwv[r] * nz;
        }
    }
}

// ---------------------------------------------------------------------------
extern "C" void kernel_launch(void* const* d_in, const int* in_sizes, int n_in,
                              void* d_out, int out_size, void* d_ws, size_t ws_size,
                              hipStream_t stream) {
    const float* x        = (const float*)d_in[0];
    const float* style    = (const float*)d_in[1];
    const float* noise    = (const float*)d_in[2];
    const float* weight   = (const float*)d_in[3];
    const float* mod_w    = (const float*)d_in[4];
    const float* noise_wt = (const float*)d_in[5];
    float* out = (float*)d_out;

    float*  s     = (float*)d_ws;                    // B*CIN
    float*  demod = s + BATCH * CIN;                 // B*COUT
    __bf16* wsW   = (__bf16*)(demod + BATCH * COUT); // 4.5 MB of A-fragments
    __bf16* xbf   = wsW + (size_t)BATCH * MT * NCHUNK * 512; // 67 MB NHWC bf16

    const size_t need = (size_t)16384 + (size_t)BATCH * MT * NCHUNK * 512 * 2
                      + (size_t)BATCH * HH * WW * CIN * 2;
    const bool big = ws_size >= need;

    k_style<<<dim3(BATCH), dim3(CIN), 0, stream>>>(style, mod_w, s);
    k_demod<<<dim3(BATCH * COUT), dim3(128), 0, stream>>>(weight, s, demod);
    k_pack <<<dim3(BATCH * MT * NCHUNK), dim3(32), 0, stream>>>(weight, s, demod, wsW);

    if (big) {
        k_cvt<<<dim3(BATCH * HH), dim3(256), 0, stream>>>(x, xbf);
        k_conv<1><<<dim3(BATCH * HH * 2), dim3(256), 0, stream>>>(x, xbf, noise, noise_wt, wsW, out);
    } else {
        k_conv<0><<<dim3(BATCH * HH * 2), dim3(256), 0, stream>>>(x, xbf, noise, noise_wt, wsW, out);
    }
}
